// PPO_GNN_LSTM_65592740545291
// MI455X (gfx1250) — matrix operand order repaired and checked by reading, not compiled, not run on hardware
//
#include <hip/hip_runtime.h>
#include <hip/hip_bf16.h>
#include <math.h>

typedef __bf16 bf16;
typedef bf16 v16bf __attribute__((ext_vector_type(16)));
typedef bf16 bf16x8 __attribute__((ext_vector_type(8)));
typedef float v8f  __attribute__((ext_vector_type(8)));

#define SEQ   64
#define NNODE 1000
#define FDIM  64
#define EDIM  128
#define HEADS 2
#define CH    64
#define NTOT  (SEQ*NNODE)          // 64000
#define EGRAPH (SEQ*NNODE*8)       // 512000 graph edges
#define ETOT  (EGRAPH + NTOT)      // + self loops = 576000
#define MPAD  1008                 // 63*16, padded LSTM batch

// ---------------------------------------------------------------------------
// WMMA GEMM: C[M,N] = A[M,K](bf16 row-major) @ B(pre-packed fragments) (+bias)(+=C)
// One wave computes a 16x64 strip (4 n-tiles). M mult of 16, N mult of 64,
// K in {64,128} (templated, fully unrolled).
//
// A fragment (16x32 bf16, contiguous k order, same as B packing): lane l, g=l>>4, m=l&15;
//   a[0..7]  = A[row, k0+16g+0 .. +7]  (one b128 load)
//   a[8..15] = A[row, k0+16g+8 .. +15] (one b128 load)
// B packed layout: for k-tile kt, n-tile tn, lane l, slot j(0..15):
//   Bp[(((kt*ntiles)+tn)*32 + l)*16 + j] = B[kt*32 + 16*(l>>4) + j][tn*16 + (l&15)]
//   -> per-lane fragment is one contiguous 32B load (2 x b128).
// C/D (16x16 f32): reg i -> M = i + 8g, N = lane&15
// ---------------------------------------------------------------------------
template<int K, int FLAGS /*1=accumulate*/>
__global__ __launch_bounds__(256)
void gemm_bf16_wmma(const bf16* __restrict__ A, const bf16* __restrict__ Bp,
                    float* __restrict__ Cm, const float* __restrict__ bias,
                    int M, int Nn) {
    int wave = (blockIdx.x * blockDim.x + threadIdx.x) >> 5;
    int lane = threadIdx.x & 31;
    int nquad = Nn >> 6;                   // 4 tiles per wave along N
    int total = (M >> 4) * nquad;
    if (wave >= total) return;             // wave-uniform: EXEC all-ones inside
    int tm = wave / nquad, tq = wave - tm * nquad;
    int g = lane >> 4, r = lane & 15;
    int ntiles = Nn >> 4;

    v8f acc[4];
#pragma unroll
    for (int j = 0; j < 4; ++j)
#pragma unroll
        for (int i = 0; i < 8; ++i) acc[j][i] = 0.f;

    const bf16* Arow = A + (size_t)(tm * 16 + r) * K + 16 * g;

#pragma unroll
    for (int kt = 0; kt < K / 32; ++kt) {
        bf16x8 a0 = *(const bf16x8*)(Arow + kt * 32);        // b128
        bf16x8 a1 = *(const bf16x8*)(Arow + kt * 32 + 8);    // b128
        v16bf a = __builtin_shufflevector(a0, a1,
                    0,1,2,3,4,5,6,7,8,9,10,11,12,13,14,15);
#pragma unroll
        for (int j = 0; j < 4; ++j) {
            int tn = tq * 4 + j;
            v16bf b = *(const v16bf*)(Bp +
                        (((size_t)kt * ntiles + tn) * 32 + lane) * 16); // 2 x b128
            acc[j] = __builtin_amdgcn_wmma_f32_16x16x32_bf16(
                         false, a, false, b, (short)0, acc[j], false, false);
        }
    }

#pragma unroll
    for (int j = 0; j < 4; ++j) {
        int n = (tq * 4 + j) * 16 + r;
        float bv = bias ? bias[n] : 0.f;
#pragma unroll
        for (int i = 0; i < 8; ++i) {
            int m = tm * 16 + i + 8 * g;
            size_t idx = (size_t)m * Nn + n;
            float v = acc[j][i] + bv;
            if (FLAGS & 1) v += Cm[idx];
            Cm[idx] = v;
        }
    }
}

// Pack f32 weight matrix [K x N] (or transposed source [N x K]) into
// fragment-ordered bf16 buffer consumed by gemm_bf16_wmma.
__global__ void pack_b_k(const float* __restrict__ src, bf16* __restrict__ dst,
                         int K, int N, int transposed) {
    int t = blockIdx.x * blockDim.x + threadIdx.x;
    if (t >= K * N) return;
    int j = t & 15, lane = (t >> 4) & 31;
    int rest = t >> 9;
    int ntiles = N >> 4;
    int tn = rest % ntiles, kt = rest / ntiles;
    int k = kt * 32 + ((lane >> 4) << 4) + j;
    int n = tn * 16 + (lane & 15);
    float v = transposed ? src[(size_t)n * K + k] : src[(size_t)k * N + n];
    dst[t] = (bf16)v;
}

// ----------------------------- small utilities -----------------------------
__global__ void f32_to_bf16_k(const float* __restrict__ s, bf16* __restrict__ d, int n) {
    int t = blockIdx.x * blockDim.x + threadIdx.x;
    if (t < n) d[t] = (bf16)s[t];
}
__global__ void fill_f32_k(float* p, float v, int n) {
    int t = blockIdx.x * blockDim.x + threadIdx.x;
    if (t < n) p[t] = v;
}
__global__ void fill_bf16_zero_k(bf16* p, int n) {
    int t = blockIdx.x * blockDim.x + threadIdx.x;
    if (t < n) p[t] = (bf16)0.0f;
}
__global__ void copy_f32_k(const float* s, float* d, int n) {
    int t = blockIdx.x * blockDim.x + threadIdx.x;
    if (t < n) d[t] = s[t];
}
__global__ void vec_add_k(const float* a, const float* b, float* d, int n) {
    int t = blockIdx.x * blockDim.x + threadIdx.x;
    if (t < n) d[t] = a[t] + b[t];
}
__global__ void add_bias_relu_k(float* x, const float* __restrict__ bias, int rows, int relu) {
    int t = blockIdx.x * blockDim.x + threadIdx.x;
    if (t >= rows * EDIM) return;
    float v = x[t] + bias[t & (EDIM - 1)];
    if (relu) v = v > 0.f ? v : 0.f;
    x[t] = v;
}

// ----------------------------- GAT attention -------------------------------
__device__ inline void atomicMaxFloat(float* addr, float val) {
    if (val >= 0.f) atomicMax((int*)addr, __float_as_int(val));
    else            atomicMin((unsigned int*)addr, __float_as_uint(val));
}
__device__ inline void edge_endpoints(const int* __restrict__ ei, int e, int& s, int& d) {
    if (e < EGRAPH) { s = ei[e]; d = ei[EGRAPH + e]; }
    else            { s = d = e - EGRAPH; }   // self loops
}
__global__ void edge_logit_k(const float* __restrict__ xl, const int* __restrict__ ei,
                             const float* __restrict__ att, float* __restrict__ logit,
                             float* __restrict__ segm) {
    int t = blockIdx.x * blockDim.x + threadIdx.x;
    if (t >= ETOT * HEADS) return;
    int e = t >> 1, h = t & 1, sN, dN;
    edge_endpoints(ei, e, sN, dN);
    const float* xs = xl + (size_t)sN * EDIM + h * CH;
    const float* xd = xl + (size_t)dN * EDIM + h * CH;
    const float* ah = att + h * CH;
    float acc = 0.f;
#pragma unroll 4
    for (int c = 0; c < CH; ++c) {
        float v = xs[c] + xd[c];
        v = v > 0.f ? v : 0.2f * v;           // leaky relu 0.2
        acc += v * ah[c];
    }
    logit[t] = acc;
    atomicMaxFloat(&segm[dN * HEADS + h], acc);
}
__global__ void edge_exp_k(float* __restrict__ logit, const int* __restrict__ ei,
                           const float* __restrict__ segm, float* __restrict__ segd) {
    int t = blockIdx.x * blockDim.x + threadIdx.x;
    if (t >= ETOT * HEADS) return;
    int e = t >> 1, h = t & 1, sN, dN;
    edge_endpoints(ei, e, sN, dN);
    float a = __expf(logit[t] - segm[dN * HEADS + h]);
    logit[t] = a;                              // reuse buffer for 'a'
    atomicAdd(&segd[dN * HEADS + h], a);
}
__global__ void edge_scatter_k(const float* __restrict__ xl, const int* __restrict__ ei,
                               const float* __restrict__ aArr, const float* __restrict__ segd,
                               float* __restrict__ out) {
    int t = blockIdx.x * blockDim.x + threadIdx.x;
    if (t >= ETOT * HEADS) return;
    int e = t >> 1, h = t & 1, sN, dN;
    edge_endpoints(ei, e, sN, dN);
    float alpha = aArr[t] / segd[dN * HEADS + h];
    const float* xs = xl + (size_t)sN * EDIM + h * CH;
    float* od = out + (size_t)dN * EDIM + h * CH;
#pragma unroll 4
    for (int c = 0; c < CH; ++c) atomicAdd(&od[c], xs[c] * alpha);
}

// --------------------------------- LSTM ------------------------------------
__global__ void lstm_point_k(const float* __restrict__ gates, float* __restrict__ cst,
                             float* __restrict__ mu, bf16* __restrict__ hbf,
                             float* __restrict__ hFin, float* __restrict__ cFin, int s) {
    int t = blockIdx.x * blockDim.x + threadIdx.x;
    if (t >= NNODE * EDIM) return;
    int n = t >> 7, e = t & 127;
    const float* g = gates + (size_t)n * (4 * EDIM);
    float i_ = g[e], f_ = g[EDIM + e], gg = g[2 * EDIM + e], o_ = g[3 * EDIM + e];
    float si = 1.f / (1.f + __expf(-i_));
    float sf = 1.f / (1.f + __expf(-f_));
    float so = 1.f / (1.f + __expf(-o_));
    float c = sf * cst[t] + si * tanhf(gg);
    float h = so * tanhf(c);
    cst[t] = c;
    mu[(size_t)(s * NNODE + n) * EDIM + e] = h;
    hbf[(size_t)n * EDIM + e] = (bf16)h;
    if (s == SEQ - 1) { hFin[t] = h; cFin[t] = c; }
}

// --------------------------------- head ------------------------------------
__global__ void meanpool_k(const float* __restrict__ mu, float* __restrict__ mp) {
    int s = blockIdx.x, e = threadIdx.x;     // 64 blocks x 128 threads, coalesced
    float acc = 0.f;
    for (int n = 0; n < NNODE; ++n) acc += mu[(size_t)(s * NNODE + n) * EDIM + e];
    mp[s * EDIM + e] = acc * (1.0f / NNODE);
}
__global__ void head_logits_k(const float* __restrict__ gs, const float* __restrict__ la,
                              const float* __restrict__ w5, const float* __restrict__ b5,
                              float* __restrict__ lg) {
    int t = blockIdx.x * blockDim.x + threadIdx.x;
    if (t >= NTOT) return;
    int s = t / NNODE;
    const float* gr = gs + s * EDIM;
    const float* lr = la + (size_t)t * EDIM;
    float acc = b5[0];
#pragma unroll 4
    for (int e = 0; e < EDIM; ++e) {
        float a1 = gr[e]; a1 = a1 > 0.f ? a1 : 0.f;
        float a2 = lr[e]; a2 = a2 > 0.f ? a2 : 0.f;
        acc += a1 * w5[e] + a2 * w5[EDIM + e];
    }
    lg[t] = acc;
}
__global__ __launch_bounds__(256)
void softmax_masked_k(const float* __restrict__ lg, const unsigned char* __restrict__ reach,
                      float* __restrict__ prob) {
    int s = blockIdx.x, tid = threadIdx.x;
    __shared__ float red[256];
    float mx = -INFINITY;
    for (int n = tid; n < NNODE; n += 256)
        if (reach[s * NNODE + n]) mx = fmaxf(mx, lg[s * NNODE + n]);
    red[tid] = mx; __syncthreads();
    for (int o = 128; o > 0; o >>= 1) { if (tid < o) red[tid] = fmaxf(red[tid], red[tid + o]); __syncthreads(); }
    mx = red[0]; __syncthreads();
    float sum = 0.f;
    for (int n = tid; n < NNODE; n += 256)
        if (reach[s * NNODE + n]) sum += __expf(lg[s * NNODE + n] - mx);
    red[tid] = sum; __syncthreads();
    for (int o = 128; o > 0; o >>= 1) { if (tid < o) red[tid] += red[tid + o]; __syncthreads(); }
    float inv = 1.f / red[0];
    for (int n = tid; n < NNODE; n += 256) {
        float v = reach[s * NNODE + n] ? __expf(lg[s * NNODE + n] - mx) * inv : 0.f;
        prob[s * NNODE + n] = v;
    }
}

// ------------------------------- host side ---------------------------------
static inline dim3 g1(long n) { return dim3((unsigned)((n + 255) / 256)); }

static void launch_gemm(const bf16* A, const bf16* Bp, float* Cm, const float* bias,
                        int M, int Nn, int K, int accumulate, hipStream_t stream) {
    int waves = (M / 16) * (Nn / 64);
    int blocks = (waves + 7) / 8;
    if (K == 64)
        gemm_bf16_wmma<64, 0><<<blocks, 256, 0, stream>>>(A, Bp, Cm, bias, M, Nn);
    else if (accumulate)
        gemm_bf16_wmma<128, 1><<<blocks, 256, 0, stream>>>(A, Bp, Cm, bias, M, Nn);
    else
        gemm_bf16_wmma<128, 0><<<blocks, 256, 0, stream>>>(A, Bp, Cm, bias, M, Nn);
}

static void run_attention(const float* xl, const int* ei, const float* att,
                          float* logit, float* segm, float* segd, float* out,
                          hipStream_t stream) {
    fill_f32_k<<<g1(NTOT * HEADS), 256, 0, stream>>>(segm, -INFINITY, NTOT * HEADS);
    fill_f32_k<<<g1(NTOT * HEADS), 256, 0, stream>>>(segd, 0.f, NTOT * HEADS);
    fill_f32_k<<<g1(NTOT * EDIM), 256, 0, stream>>>(out, 0.f, NTOT * EDIM);
    edge_logit_k<<<g1((long)ETOT * HEADS), 256, 0, stream>>>(xl, ei, att, logit, segm);
    edge_exp_k<<<g1((long)ETOT * HEADS), 256, 0, stream>>>(logit, ei, segm, segd);
    edge_scatter_k<<<g1((long)ETOT * HEADS), 256, 0, stream>>>(xl, ei, logit, segd, out);
}

extern "C" void kernel_launch(void* const* d_in, const int* in_sizes, int n_in,
                              void* d_out, int out_size, void* d_ws, size_t ws_size,
                              hipStream_t stream) {
    (void)in_sizes; (void)n_in; (void)out_size; (void)ws_size;
    const float* nfm    = (const float*)d_in[0];
    const int*   ei     = (const int*)d_in[1];
    const unsigned char* reach = (const unsigned char*)d_in[2];
    const float* h0     = (const float*)d_in[3];
    const float* c0     = (const float*)d_in[4];
    const float* gW1    = (const float*)d_in[5];
    const float* gb1    = (const float*)d_in[6];
    const float* gatt1  = (const float*)d_in[7];
    const float* gbias1 = (const float*)d_in[8];
    const float* gW     = (const float*)d_in[9];
    const float* gb     = (const float*)d_in[10];
    const float* gatt   = (const float*)d_in[11];
    const float* gbias  = (const float*)d_in[12];
    const float* Wih    = (const float*)d_in[13];
    const float* Whh    = (const float*)d_in[14];
    const float* bih    = (const float*)d_in[15];
    const float* bhh    = (const float*)d_in[16];
    const float* th6W   = (const float*)d_in[17];
    const float* th6b   = (const float*)d_in[18];
    const float* th7W   = (const float*)d_in[19];
    const float* th7b   = (const float*)d_in[20];
    const float* th5W   = (const float*)d_in[21];
    const float* th5b   = (const float*)d_in[22];

    float* out_prob = (float*)d_out;                 // [S*N]
    float* out_h    = (float*)d_out + NTOT;          // [N*E]
    float* out_c    = (float*)d_out + NTOT + NNODE * EDIM;

    char* w = (char*)d_ws;
    auto alloc = [&](size_t bytes) { void* p = (void*)w; w += (bytes + 255) & ~(size_t)255; return p; };
    float* bufA  = (float*)alloc((size_t)NTOT * EDIM * 4);
    float* bufB  = (float*)alloc((size_t)NTOT * EDIM * 4);
    float* bufC  = (float*)alloc((size_t)NTOT * EDIM * 4);
    bf16*  bufBF = (bf16*)alloc((size_t)(NTOT + 16) * EDIM * 2); // +16 rows slack for padded LSTM GEMM
    float* logit = (float*)alloc((size_t)ETOT * HEADS * 4);
    float* segm  = (float*)alloc((size_t)NTOT * HEADS * 4);
    float* segd  = (float*)alloc((size_t)NTOT * HEADS * 4);
    float* gates = (float*)alloc((size_t)MPAD * 4 * EDIM * 4);
    float* cst   = (float*)alloc((size_t)MPAD * EDIM * 4);
    bf16*  hbf   = (bf16*)alloc((size_t)MPAD * EDIM * 2);
    float* mp    = (float*)alloc((size_t)SEQ * EDIM * 4);
    bf16*  mpBF  = (bf16*)alloc((size_t)SEQ * EDIM * 2);
    float* gsBuf = (float*)alloc((size_t)SEQ * EDIM * 4);
    float* lgBuf = (float*)alloc((size_t)NTOT * 4);
    bf16* wW1   = (bf16*)alloc((size_t)FDIM * EDIM * 2);
    bf16* wWg   = (bf16*)alloc((size_t)4 * EDIM * EDIM * 2);
    bf16* wWihT = (bf16*)alloc((size_t)EDIM * 4 * EDIM * 2);
    bf16* wWhhT = (bf16*)alloc((size_t)EDIM * 4 * EDIM * 2);
    bf16* wT6   = (bf16*)alloc((size_t)EDIM * EDIM * 2);
    bf16* wT7   = (bf16*)alloc((size_t)EDIM * EDIM * 2);
    float* bb   = (float*)alloc((size_t)4 * EDIM * 4);

    // ---- weight prep: pack into WMMA fragment order (+bf16, +transpose) ----
    pack_b_k<<<g1(FDIM * EDIM), 256, 0, stream>>>(gW1, wW1, FDIM, EDIM, 0);
    for (int i = 0; i < 4; ++i)
        pack_b_k<<<g1(EDIM * EDIM), 256, 0, stream>>>(gW + (size_t)i * EDIM * EDIM,
                                                      wWg + (size_t)i * EDIM * EDIM,
                                                      EDIM, EDIM, 0);
    pack_b_k<<<g1(EDIM * 4 * EDIM), 256, 0, stream>>>(Wih, wWihT, EDIM, 4 * EDIM, 1);
    pack_b_k<<<g1(EDIM * 4 * EDIM), 256, 0, stream>>>(Whh, wWhhT, EDIM, 4 * EDIM, 1);
    pack_b_k<<<g1(EDIM * EDIM), 256, 0, stream>>>(th6W, wT6, EDIM, EDIM, 0);
    pack_b_k<<<g1(EDIM * EDIM), 256, 0, stream>>>(th7W, wT7, EDIM, EDIM, 0);
    vec_add_k<<<g1(4 * EDIM), 256, 0, stream>>>(bih, bhh, bb, 4 * EDIM);
    fill_bf16_zero_k<<<g1(16 * EDIM), 256, 0, stream>>>(bufBF + (size_t)NTOT * EDIM, 16 * EDIM);

    // ---- GAT layer 1 (K = F = 64) ----
    f32_to_bf16_k<<<g1(NTOT * FDIM), 256, 0, stream>>>(nfm, bufBF, NTOT * FDIM);
    launch_gemm(bufBF, wW1, bufA, gb1, NTOT, EDIM, FDIM, 0, stream);
    run_attention(bufA, ei, gatt1, logit, segm, segd, bufB, stream);
    add_bias_relu_k<<<g1(NTOT * EDIM), 256, 0, stream>>>(bufB, gbias1, NTOT, 1);

    // ---- GAT layers 2..5 (K = E = 128), rotate B/C; xl always in bufA ----
    float* x = bufB; float* other = bufC;
    for (int i = 0; i < 4; ++i) {
        f32_to_bf16_k<<<g1(NTOT * EDIM), 256, 0, stream>>>(x, bufBF, NTOT * EDIM);
        launch_gemm(bufBF, wWg + (size_t)i * EDIM * EDIM, bufA, gb + i * EDIM,
                    NTOT, EDIM, EDIM, 0, stream);
        run_attention(bufA, ei, gatt + i * HEADS * CH, logit, segm, segd, other, stream);
        add_bias_relu_k<<<g1(NTOT * EDIM), 256, 0, stream>>>(other, gbias + i * EDIM,
                                                             NTOT, (i < 3) ? 1 : 0);
        float* t = x; x = other; other = t;
    }
    float* muBuf = other;   // free big buffer; x holds final GAT output

    // ---- LSTM ----
    f32_to_bf16_k<<<g1(NTOT * EDIM), 256, 0, stream>>>(x, bufBF, NTOT * EDIM);
    fill_bf16_zero_k<<<g1(MPAD * EDIM), 256, 0, stream>>>(hbf, MPAD * EDIM);
    f32_to_bf16_k<<<g1(NNODE * EDIM), 256, 0, stream>>>(h0, hbf, NNODE * EDIM);
    fill_f32_k<<<g1(MPAD * EDIM), 256, 0, stream>>>(cst, 0.f, MPAD * EDIM);
    copy_f32_k<<<g1(NNODE * EDIM), 256, 0, stream>>>(c0, cst, NNODE * EDIM);
    for (int s = 0; s < SEQ; ++s) {
        launch_gemm(bufBF + (size_t)s * NNODE * EDIM, wWihT, gates, bb,
                    MPAD, 4 * EDIM, EDIM, 0, stream);                 // x_t @ Wih^T + (bih+bhh)
        launch_gemm(hbf, wWhhT, gates, nullptr, MPAD, 4 * EDIM, EDIM, 1, stream); // += h @ Whh^T
        lstm_point_k<<<g1(NNODE * EDIM), 256, 0, stream>>>(gates, cst, muBuf, hbf,
                                                           out_h, out_c, s);
    }

    // ---- head ----
    meanpool_k<<<SEQ, EDIM, 0, stream>>>(muBuf, mp);
    f32_to_bf16_k<<<g1(SEQ * EDIM), 256, 0, stream>>>(mp, mpBF, SEQ * EDIM);
    launch_gemm(mpBF, wT6, gsBuf, th6b, SEQ, EDIM, EDIM, 0, stream);
    f32_to_bf16_k<<<g1(NTOT * EDIM), 256, 0, stream>>>(muBuf, bufBF, NTOT * EDIM);
    launch_gemm(bufBF, wT7, bufA, th7b, NTOT, EDIM, EDIM, 0, stream);
    head_logits_k<<<g1(NTOT), 256, 0, stream>>>(gsBuf, bufA, th5W, th5b, lgBuf);
    softmax_masked_k<<<SEQ, 256, 0, stream>>>(lgBuf, reach, out_prob);
}
